// Variable_Self_Attn_77120432767127
// MI455X (gfx1250) — matrix operand order repaired and checked by reading, compiled, not run on hardware
//
#include <hip/hip_runtime.h>
#include <stdint.h>

#define BATCH 32
#define GRP   8
#define SEQ   512
#define CH    256
#define BG    (BATCH*GRP)

typedef __attribute__((ext_vector_type(16))) _Float16 v16h;
typedef __attribute__((ext_vector_type(8)))  float    v8f;

union FragH { v16h h; uint32_t u[8]; };
union FragF { v8f  f; float    x[8]; };

// ---- CDNA5 wave32 WMMA fragment builders (16x16x32 f16, f32 accum) ----
// A (16xK): lane m = L%16, K-half = L/16; v0..3 -> K 0..15, v4..7 -> K 16..31.
__device__ __forceinline__ v16h ldsA_frag(const _Float16* base, int stride, int lane) {
  FragH f;
  int m = lane & 15, grp = lane >> 4;
  const _Float16* row = base + m * stride;
#pragma unroll
  for (int v = 0; v < 8; ++v) {
    int k = ((v >> 2) << 4) + grp * 8 + ((v & 3) << 1);
    f.u[v] = *(const uint32_t*)(row + k);
  }
  return f.h;
}
// B (Kx16): lane n = L%16, K-half = L/16; v0..3 -> K 0..15, v4..7 -> K 16..31; LDS tile is [N][K].
__device__ __forceinline__ v16h ldsB_frag(const _Float16* base, int stride, int lane) {
  FragH f;
  int n = lane & 15, grp = lane >> 4;
  const _Float16* row = base + n * stride;
#pragma unroll
  for (int v = 0; v < 8; ++v)
    f.u[v] = *(const uint32_t*)(row + ((v >> 2) << 4) + grp * 8 + ((v & 3) << 1));
  return f.h;
}

__device__ __forceinline__ v8f wmma_f16(v16h a, v16h b, v8f c) {
  return __builtin_amdgcn_wmma_f32_16x16x32_f16(false, a, false, b,
                                                (short)0, c, false, false);
}

__device__ __forceinline__ uint32_t pack2h(float a, float b) {
  union { _Float16 h[2]; uint32_t u; } p;
  p.h[0] = (_Float16)a; p.h[1] = (_Float16)b;
  return p.u;
}
__device__ __forceinline__ uint16_t h2u(_Float16 h) {
  union { _Float16 h; uint16_t u; } c; c.h = h; return c.u;
}

// ---- Kernel 0: W f32 -> f16 (row-major W[d][c] IS the B layout for y = x W^T) ----
__global__ void w_to_half(const float* __restrict__ w, uint16_t* __restrict__ wh) {
  int i = blockIdx.x * blockDim.x + threadIdx.x;
  wh[i] = h2u((_Float16)w[i]);
}

__global__ void zero_f32(float* __restrict__ p, int n) {
  int i = blockIdx.x * blockDim.x + threadIdx.x;
  if (i < n) p[i] = 0.0f;
}

// ---- Kernel 1: Q,K,V = x @ W^T per (bg, 128-row tile).
// LDS: xA[128][256] f16 (64KB) + wB[256][256] f16 (128KB) = 192KB.
// Each wave owns 32x32 macro-tiles (2x2 WMMA): A/B fragments reused 2x each. ----
__global__ __launch_bounds__(256)
void qkv_kernel(const float* __restrict__ x,
                const uint16_t* __restrict__ whq,
                const uint16_t* __restrict__ whk,
                const uint16_t* __restrict__ whv,
                uint16_t* __restrict__ Qg,
                uint16_t* __restrict__ Kg,
                uint16_t* __restrict__ Vtg) {
  extern __shared__ char lds[];
  _Float16* xA = (_Float16*)lds;               // [128][256]
  _Float16* wB = (_Float16*)(lds + 65536);     // [256][256]
  int bg = blockIdx.y;
  int n0 = blockIdx.x * 128;
  int tid = threadIdx.x, lane = tid & 31, wave = tid >> 5;

  const float* xt = x + ((size_t)bg * SEQ + n0) * CH;
  for (int i = tid; i < (128 * 256) / 4; i += 256) {
    float4 v = ((const float4*)xt)[i];
    uint2 st; st.x = pack2h(v.x, v.y); st.y = pack2h(v.z, v.w);
    ((uint2*)xA)[i] = st;
  }

  const uint16_t* Ws[3] = { whq, whk, whv };
  size_t base = (size_t)bg * SEQ * CH;
  int nn = lane & 15, grp = lane >> 4;
  for (int wsel = 0; wsel < 3; ++wsel) {
    // prefetch next weight matrix while this one is being consumed
    if (wsel < 2)
      __builtin_prefetch((const char*)Ws[wsel + 1] + tid * 512, 0, 0);
    __syncthreads();                                  // xA ready / wB free
    const uint4* src = (const uint4*)Ws[wsel];
    for (int i = tid; i < (256 * 256) / 8; i += 256) ((uint4*)wB)[i] = src[i];
    __syncthreads();
    // 4x8 macro-tiles of 32x32 -> 4 per wave
    for (int mac = wave; mac < 32; mac += 8) {
      int mrow = mac >> 3, ncol = mac & 7;            // 32-row / 32-col macro
      const _Float16* aBase = xA + (mrow * 32) * 256;
      const _Float16* bBase = wB + (ncol * 32) * 256;
      v8f a00 = {}, a01 = {}, a10 = {}, a11 = {};
#pragma unroll
      for (int kk = 0; kk < 256; kk += 32) {
        v16h fa0 = ldsA_frag(aBase + kk,            256, lane);
        v16h fa1 = ldsA_frag(aBase + 16 * 256 + kk, 256, lane);
        v16h fb0 = ldsB_frag(bBase + kk,            256, lane);
        v16h fb1 = ldsB_frag(bBase + 16 * 256 + kk, 256, lane);
        a00 = wmma_f16(fa0, fb0, a00);
        a01 = wmma_f16(fa0, fb1, a01);
        a10 = wmma_f16(fa1, fb0, a10);
        a11 = wmma_f16(fa1, fb1, a11);
      }
      FragF ff[4]; ff[0].f = a00; ff[1].f = a01; ff[2].f = a10; ff[3].f = a11;
#pragma unroll
      for (int q = 0; q < 4; ++q) {
        int row0 = n0 + mrow * 32 + (q >> 1) * 16;
        int dcol = ncol * 32 + (q & 1) * 16 + nn;
#pragma unroll
        for (int v = 0; v < 8; ++v) {
          int row = row0 + v + 8 * grp;               // D: m = v + 8*(lane>=16)
          uint16_t hu = h2u((_Float16)ff[q].x[v]);
          if (wsel == 0)      Qg[base + (size_t)row * CH + dcol] = hu;
          else if (wsel == 1) Kg[base + (size_t)row * CH + dcol] = hu;
          else                Vtg[base + (size_t)dcol * SEQ + row] = hu;  // transposed
        }
      }
    }
  }
}

// ---- Kernel 2: attention per (bg, 64-query tile).
// Phase A: S = Q K^T, keys staged 128 at a time (qA 32KB + kB 64KB + S 128KB = 224KB).
// Phase B: softmax + alpha-mean atomics + alpha->f16 (reusing dead staging LDS).
// Phase C: out = gamma*(alpha V) + x, V staged 128 channels at a time. ----
__global__ __launch_bounds__(256)
void attn_kernel(const float* __restrict__ x,
                 const uint16_t* __restrict__ Qg,
                 const uint16_t* __restrict__ Kg,
                 const uint16_t* __restrict__ Vtg,
                 const float* __restrict__ gamma,
                 float* __restrict__ outg,
                 float* __restrict__ alphaAcc) {
  extern __shared__ char lds[];
  _Float16* qA = (_Float16*)lds;               // [64][256]   32KB (phase A)
  _Float16* kB = (_Float16*)(lds + 32768);     // [128][256]  64KB (phase A)
  float*    S  = (float*)(lds + 98304);        // [64][512]  128KB (phase A/B)
  _Float16* aH = (_Float16*)lds;               // [64][512]   64KB (phase B/C)
  _Float16* vB = (_Float16*)(lds + 65536);     // [128][512] 128KB (phase C)

  int bg = blockIdx.y;
  int q0 = blockIdx.x * 64;
  int tid = threadIdx.x, lane = tid & 31, wave = tid >> 5;
  int nn = lane & 15, grp = lane >> 4;
  size_t base = (size_t)bg * SEQ * CH;

  const uint4* qsrc = (const uint4*)(Qg + base + (size_t)q0 * CH);
  for (int i = tid; i < (64 * 256) / 8; i += 256) ((uint4*)qA)[i] = qsrc[i];

  // ---- Phase A: scores, 128 keys per chunk ----
  for (int kc = 0; kc < 4; ++kc) {
    if (kc < 3)
      __builtin_prefetch((const char*)(Kg + base + (size_t)(kc + 1) * 128 * CH)
                             + tid * 256, 0, 0);
    __syncthreads();
    const uint4* ksrc = (const uint4*)(Kg + base + (size_t)kc * 128 * CH);
    for (int i = tid; i < (128 * 256) / 8; i += 256) ((uint4*)kB)[i] = ksrc[i];
    __syncthreads();
    // 2x4 macro-tiles of 32x32 -> 1 per wave
    {
      int mrow = wave >> 2, ncol = wave & 3;
      const _Float16* aBase = qA + (mrow * 32) * 256;
      const _Float16* bBase = kB + (ncol * 32) * 256;
      v8f a00 = {}, a01 = {}, a10 = {}, a11 = {};
#pragma unroll
      for (int kk = 0; kk < 256; kk += 32) {
        v16h fa0 = ldsA_frag(aBase + kk,            256, lane);
        v16h fa1 = ldsA_frag(aBase + 16 * 256 + kk, 256, lane);
        v16h fb0 = ldsB_frag(bBase + kk,            256, lane);
        v16h fb1 = ldsB_frag(bBase + 16 * 256 + kk, 256, lane);
        a00 = wmma_f16(fa0, fb0, a00);
        a01 = wmma_f16(fa0, fb1, a01);
        a10 = wmma_f16(fa1, fb0, a10);
        a11 = wmma_f16(fa1, fb1, a11);
      }
      FragF ff[4]; ff[0].f = a00; ff[1].f = a01; ff[2].f = a10; ff[3].f = a11;
#pragma unroll
      for (int q = 0; q < 4; ++q) {
        int m0 = mrow * 32 + (q >> 1) * 16;
        int col = kc * 128 + ncol * 32 + (q & 1) * 16 + nn;
#pragma unroll
        for (int v = 0; v < 8; ++v)
          S[(m0 + v + 8 * grp) * 512 + col] = ff[q].x[v];
      }
    }
  }
  __syncthreads();

  // ---- Phase B: softmax (4 consecutive lanes per row) + alpha-mean atomics ----
  {
    int r = tid >> 2, sub = tid & 3;
    float* srow = S + r * 512;
    float mx = -3.0e38f;
    for (int c = sub; c < 512; c += 4) mx = fmaxf(mx, srow[c]);
    mx = fmaxf(mx, __shfl_xor(mx, 1));
    mx = fmaxf(mx, __shfl_xor(mx, 2));
    float sum = 0.0f;
    for (int c = sub; c < 512; c += 4) {
      float e = __expf(srow[c] - mx);
      srow[c] = e;
      sum += e;
    }
    sum += __shfl_xor(sum, 1);
    sum += __shfl_xor(sum, 2);
    float inv = 1.0f / sum;
    float* arow = alphaAcc + (size_t)(q0 + r) * SEQ;
    for (int c = sub; c < 512; c += 4) {
      float a = srow[c] * inv;
      aH[r * 512 + c] = (_Float16)a;                       // alpha f16 for PV
      atomicAdd(arow + c, a * (1.0f / (float)BG));         // mean over b,g
    }
  }

  // ---- Phase C: out = gamma * (alpha @ V) + x, 128 channels per chunk ----
  float g = gamma[0];
  for (int dc = 0; dc < 2; ++dc) {
    if (dc < 1)
      __builtin_prefetch((const char*)(Vtg + base + (size_t)128 * SEQ)
                             + tid * 512, 0, 0);
    __syncthreads();                             // aH done / vB (over S) free
    const uint4* vsrc = (const uint4*)(Vtg + base + (size_t)dc * 128 * SEQ);
    for (int i = tid; i < (128 * 512) / 8; i += 256) ((uint4*)vB)[i] = vsrc[i];
    __syncthreads();
    // 2x4 macro-tiles of 32x32 -> 1 per wave
    {
      int mrow = wave >> 2, ncol = wave & 3;
      const _Float16* aBase = aH + (mrow * 32) * 512;
      const _Float16* bBase = vB + (ncol * 32) * 512;
      v8f a00 = {}, a01 = {}, a10 = {}, a11 = {};
#pragma unroll
      for (int kk = 0; kk < 512; kk += 32) {
        v16h fa0 = ldsA_frag(aBase + kk,            512, lane);
        v16h fa1 = ldsA_frag(aBase + 16 * 512 + kk, 512, lane);
        v16h fb0 = ldsB_frag(bBase + kk,            512, lane);
        v16h fb1 = ldsB_frag(bBase + 16 * 512 + kk, 512, lane);
        a00 = wmma_f16(fa0, fb0, a00);
        a01 = wmma_f16(fa0, fb1, a01);
        a10 = wmma_f16(fa1, fb0, a10);
        a11 = wmma_f16(fa1, fb1, a11);
      }
      FragF ff[4]; ff[0].f = a00; ff[1].f = a01; ff[2].f = a10; ff[3].f = a11;
#pragma unroll
      for (int q = 0; q < 4; ++q) {
        int row0 = q0 + mrow * 32 + (q >> 1) * 16;
        int dcol = dc * 128 + ncol * 32 + (q & 1) * 16 + nn;
#pragma unroll
        for (int v = 0; v < 8; ++v) {
          int row = row0 + v + 8 * grp;
          size_t idx = base + (size_t)row * CH + dcol;
          outg[idx] = g * ff[q].x[v] + x[idx];   // fused residual
        }
      }
    }
  }
}

extern "C" void kernel_launch(void* const* d_in, const int* in_sizes, int n_in,
                              void* d_out, int out_size, void* d_ws, size_t ws_size,
                              hipStream_t stream) {
  (void)in_sizes; (void)n_in; (void)out_size; (void)ws_size;
  const float* x     = (const float*)d_in[0];
  const float* Wq    = (const float*)d_in[1];
  const float* Wk    = (const float*)d_in[2];
  const float* Wv    = (const float*)d_in[3];
  const float* gamma = (const float*)d_in[4];

  // Workspace (f16): 3 weight matrices (384KB) + Q + K + Vt (64MB each).
  char* ws = (char*)d_ws;
  const size_t WBYTES = (size_t)CH * CH * 2;          // 128KB
  const size_t TBYTES = (size_t)BG * SEQ * CH * 2;    // 64MB
  uint16_t* whq = (uint16_t*)(ws);
  uint16_t* whk = (uint16_t*)(ws + WBYTES);
  uint16_t* whv = (uint16_t*)(ws + 2 * WBYTES);
  uint16_t* Qg  = (uint16_t*)(ws + 3 * WBYTES);
  uint16_t* Kg  = (uint16_t*)(ws + 3 * WBYTES + TBYTES);
  uint16_t* Vtg = (uint16_t*)(ws + 3 * WBYTES + 2 * TBYTES);

  float* outg = (float*)d_out;
  float* alphaAcc = outg + (size_t)BG * SEQ * CH;     // alpha_mean region of d_out

  w_to_half<<<CH * CH / 256, 256, 0, stream>>>(Wq, whq);
  w_to_half<<<CH * CH / 256, 256, 0, stream>>>(Wk, whk);
  w_to_half<<<CH * CH / 256, 256, 0, stream>>>(Wv, whv);
  zero_f32<<<SEQ * SEQ / 256, 256, 0, stream>>>(alphaAcc, SEQ * SEQ);
  qkv_kernel<<<dim3(SEQ / 128, BG), 256, 196608, stream>>>(
      x, whq, whk, whv, Qg, Kg, Vtg);
  attn_kernel<<<dim3(SEQ / 64, BG), 256, 229376, stream>>>(
      x, Qg, Kg, Vtg, gamma, outg, alphaAcc);
}
